// DeepSeekMoE_68882685493800
// MI455X (gfx1250) — compile-verified, matrix operand order checked by reading, not run on hardware
//
#include <hip/hip_runtime.h>

#define T_     2048
#define DIM_   1024
#define INTER_ 1408
#define E_     16
#define S_     2
#define TOPK_  4

typedef __attribute__((ext_vector_type(8)))  float        v8f;
typedef __attribute__((ext_vector_type(16))) __bf16       v16bf;
typedef __attribute__((ext_vector_type(4)))  unsigned int u32x4;
typedef __attribute__((ext_vector_type(4)))  float        f32x4;

union Frag {
  v16bf v;
  u32x4 q[2];
};

__device__ __forceinline__ unsigned short f2bf(float f) {
  unsigned u = __float_as_uint(f);
  unsigned r = u + 0x7FFFu + ((u >> 16) & 1u);   // round-to-nearest-even
  return (unsigned short)(r >> 16);
}

__device__ __forceinline__ unsigned pack_bf2(float lo, float hi) {
  return (unsigned)f2bf(lo) | ((unsigned)f2bf(hi) << 16);
}

__device__ __forceinline__ v8f vzero8() {
  v8f z = {0.f, 0.f, 0.f, 0.f, 0.f, 0.f, 0.f, 0.f};
  return z;
}

// Async 16-byte global -> LDS copy (CDNA5 ASYNCcnt path), via inline asm so we
// don't depend on the clang builtin's exact pointer-type signature.
__device__ __forceinline__ void async_g2l_b128(const void* gp, void* lp) {
  unsigned lds_off = (unsigned)(uintptr_t)lp;            // low 32 bits = LDS offset
  unsigned long long ga = (unsigned long long)(uintptr_t)gp;
  asm volatile("global_load_async_to_lds_b128 %0, %1, off"
               :: "v"(lds_off), "v"(ga) : "memory");
}

__device__ __forceinline__ void wait_asynccnt0() {
  asm volatile("s_wait_asynccnt 0" ::: "memory");
}

// ---------------------------------------------------------------------------
// Zero the per-expert cursors.
// ---------------------------------------------------------------------------
__global__ void k_init(int* __restrict__ cnt) {
  if (threadIdx.x < E_) cnt[threadIdx.x] = 0;
}

// ---------------------------------------------------------------------------
// x fp32 -> bf16 (row-major copy), enables async LDS gather in the FFN.
// ---------------------------------------------------------------------------
__global__ __launch_bounds__(256) void k_xcast(
    const float* __restrict__ x, unsigned short* __restrict__ xb) {
  size_t i = ((size_t)blockIdx.x * 256 + threadIdx.x) * 8;
  f32x4 a = *(const f32x4*)(x + i);
  f32x4 b = *(const f32x4*)(x + i + 4);
  u32x4 o;
  o[0] = pack_bf2(a[0], a[1]);
  o[1] = pack_bf2(a[2], a[3]);
  o[2] = pack_bf2(b[0], b[1]);
  o[3] = pack_bf2(b[2], b[3]);
  *(u32x4*)(xb + i) = o;
}

// ---------------------------------------------------------------------------
// fp32 [K][N] (per expert, blockIdx.z) -> bf16 [N][K]  (transpose + convert)
// ---------------------------------------------------------------------------
__global__ __launch_bounds__(256) void k_transpose_bf16(
    const float* __restrict__ W, unsigned short* __restrict__ WT, int K, int N) {
  __shared__ unsigned short L[32][33];
  const int n0 = blockIdx.x * 32, k0 = blockIdx.y * 32;
  const size_t mat = (size_t)blockIdx.z * (size_t)K * (size_t)N;
  W  += mat;
  WT += mat;
  const int tid = threadIdx.x;
  for (int p = tid; p < 1024; p += 256) {
    int k = p >> 5, n = p & 31;
    L[k][n] = f2bf(W[(size_t)(k0 + k) * N + (n0 + n)]);
  }
  __syncthreads();
  for (int p = tid; p < 1024; p += 256) {
    int n = p >> 5, k = p & 31;
    WT[(size_t)(n0 + n) * K + (k0 + k)] = L[k][n];
  }
}

// ---------------------------------------------------------------------------
// Gate: sigmoid scores, group-limited top-4, scatter dispatch lists.
// One block per token, 8 waves x 2 experts each.
// ---------------------------------------------------------------------------
__global__ __launch_bounds__(256) void k_gate(
    const float* __restrict__ x, const float* __restrict__ gw,
    const float* __restrict__ bias, float* __restrict__ wts,
    int* __restrict__ perm, int* __restrict__ cnt) {
  const int t = blockIdx.x, tid = threadIdx.x;
  const int wave = tid >> 5, lane = tid & 31;
  __shared__ float ssc[E_];

  for (int e2 = wave * 2; e2 < wave * 2 + 2; ++e2) {
    float p = 0.f;
    for (int i = lane; i < DIM_; i += 32)
      p += x[t * DIM_ + i] * gw[e2 * DIM_ + i];
    for (int o = 16; o > 0; o >>= 1) p += __shfl_xor(p, o, 32);
    if (lane == 0) ssc[e2] = p;
  }
  __syncthreads();

  if (tid == 0) {
    float sc[E_], bi[E_];
    for (int e = 0; e < E_; ++e) {
      sc[e] = 1.f / (1.f + expf(-ssc[e]));
      bi[e] = sc[e] + bias[e];
    }
    float gs[4];
    for (int g = 0; g < 4; ++g) {
      float m1 = -1e30f, m2 = -1e30f;
      for (int j = 0; j < 4; ++j) {
        float v = bi[g * 4 + j];
        if (v > m1) { m2 = m1; m1 = v; }
        else if (v > m2) { m2 = v; }
      }
      gs[g] = m1 + m2;
    }
    int g1 = 0;
    for (int g = 1; g < 4; ++g) if (gs[g] > gs[g1]) g1 = g;
    int g2 = -1;
    for (int g = 0; g < 4; ++g)
      if (g != g1 && (g2 < 0 || gs[g] > gs[g2])) g2 = g;

    float masked[E_];
    for (int e = 0; e < E_; ++e) {
      int g = e >> 2;
      masked[e] = (g == g1 || g == g2) ? bi[e] : -1e30f;
    }
    int sel[TOPK_];
    float wsel[TOPK_], sum = 0.f;
    for (int k = 0; k < TOPK_; ++k) {
      int best = 0;
      for (int e = 1; e < E_; ++e) if (masked[e] > masked[best]) best = e;
      sel[k] = best;
      wsel[k] = sc[best];
      masked[best] = -2e30f;
      sum += wsel[k];
    }
    float inv = 1.f / fmaxf(sum, 1e-10f);
    for (int k = 0; k < TOPK_; ++k) {
      wts[t * TOPK_ + k] = wsel[k] * inv;          // ROUTE_SCALE == 1
      int e = sel[k];
      int pos = atomicAdd(&cnt[e], 1);
      perm[e * T_ + pos] = t * TOPK_ + k;          // assignment id
    }
  }
}

// ---------------------------------------------------------------------------
// Fused expert FFN: Y = (silu(Xg*W1) .* (Xg*W3)) * W2, 64 rows per block.
// Blocks [0,512): routed (expert = b>>5, tile = b&31); [512,576): shared.
// Weights are bf16, [N][K] transposed. All GEMMs via v_wmma_f32_16x16x32_bf16;
// activation rows gathered into LDS with global_load_async_to_lds_b128.
// ---------------------------------------------------------------------------
__global__ __launch_bounds__(512, 1) void k_ffn(
    const unsigned short* __restrict__ xb,
    const unsigned short* __restrict__ w1T, const unsigned short* __restrict__ w2T,
    const unsigned short* __restrict__ w3T,
    const unsigned short* __restrict__ sw1T, const unsigned short* __restrict__ sw2T,
    const unsigned short* __restrict__ sw3T,
    const int* __restrict__ cnt, const int* __restrict__ perm,
    const float* __restrict__ wts, float* __restrict__ yslots) {
  constexpr int XLD = DIM_ + 16;  // padded bf16 row stride (bank-conflict free)
  constexpr int HLD = 128 + 8;

  __shared__ unsigned short Xs[64 * XLD];
  __shared__ unsigned short Hs[64 * HLD];
  __shared__ int   s_tok[64];
  __shared__ int   s_slot[64];
  __shared__ float s_w[64];

  const int b = blockIdx.x, tid = threadIdx.x;
  const unsigned short *W1, *W2, *W3;
  int e = -1, tile, sharedIdx = -1;
  if (b < 512) {
    e = b >> 5;
    tile = b & 31;
    if (tile * 64 >= cnt[e]) return;              // block-uniform early exit
    size_t o = (size_t)e * DIM_ * INTER_;
    W1 = w1T + o; W3 = w3T + o; W2 = w2T + o;
  } else {
    sharedIdx = (b - 512) >> 5;
    tile = (b - 512) & 31;
    size_t o = (size_t)sharedIdx * DIM_ * INTER_;
    W1 = sw1T + o; W3 = sw3T + o; W2 = sw2T + o;
  }
  const bool routed = (e >= 0);

  if (tid < 64) {
    int r = tid;
    if (routed) {
      int i = tile * 64 + r;
      if (i < cnt[e]) {
        int a = perm[e * T_ + i];
        s_tok[r] = a >> 2; s_slot[r] = a & 3; s_w[r] = wts[a];
      } else {
        s_tok[r] = -1; s_slot[r] = 0; s_w[r] = 0.f;
      }
    } else {
      s_tok[r] = tile * 64 + r; s_slot[r] = TOPK_ + sharedIdx; s_w[r] = 1.f;
    }
  }
  __syncthreads();

  // Gather the 64 bf16 activation rows into LDS (async copy, 16B granules).
  for (int i = tid; i < 64 * (DIM_ / 8); i += 512) {
    int r = i >> 7, seg = i & 127;
    int tok = s_tok[r];
    if (tok < 0) tok = 0;
    const unsigned short* gp = xb + ((size_t)tok << 10) + seg * 8;
    unsigned short* lp = &Xs[r * XLD + seg * 8];
    async_g2l_b128(gp, lp);
  }
  wait_asynccnt0();
  __syncthreads();

  const int wave = tid >> 5, lane = tid & 31;
  const int ln = lane & 15, lh = lane >> 4, kb = lh * 8;
  const int mi  = wave >> 2;       // phase-A M-tile (0..3)
  const int npA = (wave & 3) * 2;  // phase-A N-tile pair within 128-col chunk
  const int n0B = wave * 64;       // phase-B output-column base

  v8f Y[4][4];
#pragma unroll
  for (int m = 0; m < 4; ++m)
#pragma unroll
    for (int n = 0; n < 4; ++n) Y[m][n] = vzero8();

#pragma unroll 1
  for (int chunk = 0; chunk < INTER_ / 128; ++chunk) {
    const int c0 = chunk * 128;
    v8f a1[2] = {vzero8(), vzero8()};
    v8f a3[2] = {vzero8(), vzero8()};

    // ---- phase A: H_chunk = silu(X*W1) .* (X*W3), K = DIM_ ----
#pragma unroll 1
    for (int k0 = 0; k0 < DIM_; k0 += 32) {
      Frag A;
      const unsigned short* xp = &Xs[(mi * 16 + ln) * XLD + k0 + kb];
      A.q[0] = *(const u32x4*)xp;
      A.q[1] = *(const u32x4*)(xp + 16);
#pragma unroll
      for (int j = 0; j < 2; ++j) {
        const int n = c0 + (npA + j) * 16 + ln;
        Frag B1;
        const unsigned short* p1 = &W1[(size_t)n * DIM_ + k0 + kb];
        B1.q[0] = *(const u32x4*)p1;
        B1.q[1] = *(const u32x4*)(p1 + 16);
        a1[j] = __builtin_amdgcn_wmma_f32_16x16x32_bf16(
            false, A.v, false, B1.v, (short)0, a1[j], false, false);
        Frag B3;
        const unsigned short* p3 = &W3[(size_t)n * DIM_ + k0 + kb];
        B3.q[0] = *(const u32x4*)p3;
        B3.q[1] = *(const u32x4*)(p3 + 16);
        a3[j] = __builtin_amdgcn_wmma_f32_16x16x32_bf16(
            false, A.v, false, B3.v, (short)0, a3[j], false, false);
      }
    }
    // silu via v_exp + v_rcp (avoid the IEEE-divide expansion)
#pragma unroll
    for (int j = 0; j < 2; ++j)
#pragma unroll
      for (int v = 0; v < 8; ++v) {
        float z = a1[j][v];
        float h = z * __builtin_amdgcn_rcpf(1.f + __expf(-z)) * a3[j][v];
        int row = mi * 16 + v + 8 * lh;
        Hs[row * HLD + (npA + j) * 16 + ln] = f2bf(h);
      }
    __syncthreads();

    // ---- phase B: Y += H_chunk * W2[chunk rows] ----
#pragma unroll 1
    for (int ks = 0; ks < 128; ks += 32) {
      Frag Am[4];
#pragma unroll
      for (int m = 0; m < 4; ++m) {
        const unsigned short* hp = &Hs[(m * 16 + ln) * HLD + ks + kb];
        Am[m].q[0] = *(const u32x4*)hp;
        Am[m].q[1] = *(const u32x4*)(hp + 16);
      }
#pragma unroll
      for (int nn = 0; nn < 4; ++nn) {
        const int n = n0B + nn * 16 + ln;
        Frag B;
        const unsigned short* p2 = &W2[(size_t)n * INTER_ + c0 + ks + kb];
        B.q[0] = *(const u32x4*)p2;
        B.q[1] = *(const u32x4*)(p2 + 16);
#pragma unroll
        for (int m = 0; m < 4; ++m)
          Y[m][nn] = __builtin_amdgcn_wmma_f32_16x16x32_bf16(
              false, Am[m].v, false, B.v, (short)0, Y[m][nn], false, false);
      }
    }
    __syncthreads();
  }

  // Weighted store into the per-(token, slot) slab; each (tok,slot) written once.
#pragma unroll
  for (int m = 0; m < 4; ++m)
#pragma unroll
    for (int v = 0; v < 8; ++v) {
      int r = m * 16 + v + 8 * lh;
      int tok = s_tok[r];
      if (tok >= 0) {
        float wv = s_w[r];
        size_t base = ((size_t)s_slot[r] * T_ + (size_t)tok) * DIM_;
#pragma unroll
        for (int nn = 0; nn < 4; ++nn)
          yslots[base + n0B + nn * 16 + ln] = Y[m][nn][v] * wv;
      }
    }
}

// ---------------------------------------------------------------------------
// out[t][d] = sum over the 6 slots (4 routed + 2 shared), fixed order.
// ---------------------------------------------------------------------------
__global__ __launch_bounds__(256) void k_combine(
    const float* __restrict__ ys, float* __restrict__ out) {
  const size_t NTOT = (size_t)T_ * DIM_;
  size_t i = ((size_t)blockIdx.x * 256 + threadIdx.x) * 4;
  f32x4 acc = *(const f32x4*)(ys + i);
#pragma unroll
  for (int s = 1; s < TOPK_ + S_; ++s) {
    f32x4 v = *(const f32x4*)(ys + (size_t)s * NTOT + i);
    acc += v;
  }
  *(f32x4*)(out + i) = acc;
}

// ---------------------------------------------------------------------------
// Workspace layout (bytes)
// ---------------------------------------------------------------------------
static constexpr size_t OFF_CNT  = 0;
static constexpr size_t OFF_WTS  = 1024;
static constexpr size_t OFF_PERM = OFF_WTS + (size_t)T_ * TOPK_ * 4;   // 33792
static constexpr size_t OFF_YS   = OFF_PERM + (size_t)E_ * T_ * 4;     // 164864
static constexpr size_t YS_BYTES = (size_t)(TOPK_ + S_) * T_ * DIM_ * 4;
static constexpr size_t WB       = (size_t)DIM_ * INTER_ * 2;          // bf16 matrix
static constexpr size_t OFF_W1T  = OFF_YS + YS_BYTES;
static constexpr size_t OFF_W3T  = OFF_W1T + (size_t)E_ * WB;
static constexpr size_t OFF_W2T  = OFF_W3T + (size_t)E_ * WB;
static constexpr size_t OFF_SW1T = OFF_W2T + (size_t)E_ * WB;
static constexpr size_t OFF_SW3T = OFF_SW1T + (size_t)S_ * WB;
static constexpr size_t OFF_SW2T = OFF_SW3T + (size_t)S_ * WB;
static constexpr size_t OFF_XBF  = OFF_SW2T + (size_t)S_ * WB;         // bf16 x

extern "C" void kernel_launch(void* const* d_in, const int* in_sizes, int n_in,
                              void* d_out, int out_size, void* d_ws, size_t ws_size,
                              hipStream_t stream) {
  const float* x    = (const float*)d_in[0];
  const float* gw   = (const float*)d_in[1];
  const float* bias = (const float*)d_in[2];
  const float* w1   = (const float*)d_in[3];
  const float* w2   = (const float*)d_in[4];
  const float* w3   = (const float*)d_in[5];
  const float* sw1  = (const float*)d_in[6];
  const float* sw2  = (const float*)d_in[7];
  const float* sw3  = (const float*)d_in[8];
  float* out = (float*)d_out;

  char* ws = (char*)d_ws;
  int*   cnt    = (int*)(ws + OFF_CNT);
  float* wts    = (float*)(ws + OFF_WTS);
  int*   perm   = (int*)(ws + OFF_PERM);
  float* yslots = (float*)(ws + OFF_YS);
  unsigned short* w1T  = (unsigned short*)(ws + OFF_W1T);
  unsigned short* w3T  = (unsigned short*)(ws + OFF_W3T);
  unsigned short* w2T  = (unsigned short*)(ws + OFF_W2T);
  unsigned short* sw1T = (unsigned short*)(ws + OFF_SW1T);
  unsigned short* sw3T = (unsigned short*)(ws + OFF_SW3T);
  unsigned short* sw2T = (unsigned short*)(ws + OFF_SW2T);
  unsigned short* xbf  = (unsigned short*)(ws + OFF_XBF);

  k_init<<<1, 32, 0, stream>>>(cnt);

  k_xcast<<<(T_ * DIM_) / (256 * 8), 256, 0, stream>>>(x, xbf);

  // One-time weight convert + transpose (fp32 [K][N] -> bf16 [N][K]).
  k_transpose_bf16<<<dim3(INTER_ / 32, DIM_ / 32, E_), 256, 0, stream>>>(w1, w1T, DIM_, INTER_);
  k_transpose_bf16<<<dim3(INTER_ / 32, DIM_ / 32, E_), 256, 0, stream>>>(w3, w3T, DIM_, INTER_);
  k_transpose_bf16<<<dim3(DIM_ / 32, INTER_ / 32, E_), 256, 0, stream>>>(w2, w2T, INTER_, DIM_);
  k_transpose_bf16<<<dim3(INTER_ / 32, DIM_ / 32, S_), 256, 0, stream>>>(sw1, sw1T, DIM_, INTER_);
  k_transpose_bf16<<<dim3(INTER_ / 32, DIM_ / 32, S_), 256, 0, stream>>>(sw3, sw3T, DIM_, INTER_);
  k_transpose_bf16<<<dim3(DIM_ / 32, INTER_ / 32, S_), 256, 0, stream>>>(sw2, sw2T, INTER_, DIM_);

  k_gate<<<T_, 256, 0, stream>>>(x, gw, bias, wts, perm, cnt);

  // 512 worst-case routed tile blocks + 64 shared-expert tile blocks.
  k_ffn<<<E_ * 32 + S_ * 32, 512, 0, stream>>>(
      xbf, w1T, w2T, w3T, sw1T, sw2T, sw3T, cnt, perm, wts, yslots);

  k_combine<<<(T_ * DIM_) / (256 * 4), 256, 0, stream>>>(yslots, out);
}
